// GraphVAE_30520037605804
// MI455X (gfx1250) — matrix operand order repaired and checked by reading, not compiled, not run on hardware
//
#include <hip/hip_runtime.h>
#include <hip/hip_bf16.h>
#include <math.h>

// ---------------------------------------------------------------------------
// MI455X (gfx1250) implementation. wave32, WMMA bf16 (16x16x32, f32 acc).
// Heavy GEMMs -> v_wmma_f32_16x16x32_bf16; activation staging for the big
// gated-fc GEMM uses global_load_async_to_lds_b128 (ASYNCcnt) from a bf16
// activation mirror; reductions/norms in f32 VALU.
// ---------------------------------------------------------------------------

typedef __bf16 bf16t;
typedef __attribute__((ext_vector_type(16))) __bf16 v16bf;
typedef __attribute__((ext_vector_type(8)))  __bf16 v8bf;
typedef __attribute__((ext_vector_type(8)))  float   v8f;

#define N_ATOMS 32768      // B*NPC
#define N_CRY   512
#define AFL     64
#define MNB     12
#define NBRF    41
#define NROWS   (N_ATOMS * MNB)   // 393216
#define KFC     169               // 2*AFL + NBR
#define KFCP    192               // padded to 6 k-tiles of 32
#define NFC     128               // 2*AFL

static __device__ inline v8f vzero8() {
  v8f a;
#pragma unroll
  for (int i = 0; i < 8; ++i) a[i] = 0.f;
  return a;
}

static __device__ inline v8f wmma_bf16(v16bf a, v16bf b, v8f c) {
  return __builtin_amdgcn_wmma_f32_16x16x32_bf16(false, a, false, b, (short)0, c,
                                                 false, false);
}

// Async copy 16B global -> LDS (per active lane). GV addressing form.
static __device__ inline void async_b128(unsigned lds_byte_addr, const void* gp) {
  asm volatile("global_load_async_to_lds_b128 %0, %1, off"
               :: "v"(lds_byte_addr), "v"(gp)
               : "memory");
}
static __device__ inline void wait_async0() {
  asm volatile("s_wait_asynccnt 0x0" ::: "memory");
}

// A fragment (16x32 bf16, row-major source, stride in elements, k0 mult of 32)
static __device__ inline v16bf load_a_frag(const bf16t* __restrict__ p0, int row0,
                                           int k0, int stride) {
  const int lane = threadIdx.x & 31;
  const int hi = lane >> 4;
  const bf16t* p = p0 + (row0 + (lane & 15)) * stride + k0 + hi * 16;
  v8bf lo = *(const v8bf*)p;
  v8bf hv = *(const v8bf*)(p + 8);
  v16bf out;
#pragma unroll
  for (int i = 0; i < 8; ++i) { out[i] = lo[i]; out[i + 8] = hv[i]; }
  return out;
}

// B fragment (32x16) from a TRANSPOSED-stored matrix L where L[n][k] = B(k,n).
static __device__ inline v16bf load_bT_frag(const bf16t* __restrict__ p0, int col0,
                                            int k0, int stride) {
  const int lane = threadIdx.x & 31;
  const int hi = lane >> 4;
  const bf16t* p = p0 + (col0 + (lane & 15)) * stride + k0 + hi * 16;
  v8bf lo = *(const v8bf*)p;
  v8bf hv = *(const v8bf*)(p + 8);
  v16bf out;
#pragma unroll
  for (int i = 0; i < 8; ++i) { out[i] = lo[i]; out[i + 8] = hv[i]; }
  return out;
}

// B fragment from pre-swizzled global storage: fragment-contiguous per lane.
static __device__ inline v16bf load_bsw(const bf16t* __restrict__ g, int kt, int nt,
                                        int NT) {
  const int lane = threadIdx.x & 31;
  const bf16t* p = g + (size_t)(((kt * NT + nt) * 32 + lane)) * 16;
  v8bf lo = *(const v8bf*)p;
  v8bf hv = *(const v8bf*)(p + 8);
  v16bf out;
#pragma unroll
  for (int i = 0; i < 8; ++i) { out[i] = lo[i]; out[i + 8] = hv[i]; }
  return out;
}

static __device__ inline float softplusf(float v) {
  return v > 20.f ? v : log1pf(expf(v));
}
static __device__ inline float sigmoidf_(float v) { return 1.f / (1.f + expf(-v)); }

// ---------------------------------------------------------------------------
// Pre-swizzle a f32 [K][Ncols] weight into WMMA B-fragment order (bf16).
// ---------------------------------------------------------------------------
__global__ void swizzleB_kernel(const float* __restrict__ src, bf16t* __restrict__ dst,
                                int K, int Ncols, int KT, int NT) {
  int t = blockIdx.x * 256 + threadIdx.x;
  int total = KT * NT * 512;
  if (t >= total) return;
  int e = t & 15;
  int lane = (t >> 4) & 31;
  int nt = (t >> 9) % NT;
  int kt = t / (512 * NT);
  int k = kt * 32 + (lane >> 4) * 16 + e;
  int n = nt * 16 + (lane & 15);
  float v = (k < K) ? src[(size_t)k * Ncols + n] : 0.f;
  dst[t] = (bf16t)v;
}

// ---------------------------------------------------------------------------
// Embedding: x = atom_fea @ emb_w + emb_b   [32768,92]x[92,64]
// ---------------------------------------------------------------------------
__global__ void __launch_bounds__(256)
embed_kernel(const float* __restrict__ af, const bf16t* __restrict__ bsw,
             const float* __restrict__ eb, float* __restrict__ x) {
  __shared__ __attribute__((aligned(32))) bf16t sA[32 * 96];
  const int tid = threadIdx.x;
  const int w = tid >> 5, lane = tid & 31, hi = lane >> 4, cl = lane & 15;
  const int rt = w >> 2, ct = w & 3;
  const int rb = blockIdx.x * 32;
  for (int e = tid; e < 32 * 96; e += 256) {
    int row = e / 96, col = e - row * 96;
    sA[e] = (bf16t)(col < 92 ? af[(size_t)(rb + row) * 92 + col] : 0.f);
  }
  __syncthreads();
  v8f acc = vzero8();
#pragma unroll
  for (int kt = 0; kt < 3; ++kt) {
    v16bf a = load_a_frag(sA, rt * 16, kt * 32, 96);
    v16bf b = load_bsw(bsw, kt, ct, 4);
    acc = wmma_bf16(a, b, acc);
  }
  const int col = ct * 16 + cl;
  const float bias = eb[col];
#pragma unroll
  for (int r = 0; r < 8; ++r) {
    int row = rt * 16 + r + 8 * hi;
    x[(size_t)(rb + row) * 64 + col] = acc[r] + bias;
  }
}

// ---------------------------------------------------------------------------
// GraphTransformer layer per crystal (one block per crystal, all in LDS).
// Writes post-LN2 activations to x (f32) and xb (bf16 mirror for fc pass).
// ---------------------------------------------------------------------------
__global__ void __launch_bounds__(256)
transformer_kernel(float* __restrict__ x, bf16t* __restrict__ xb,
                   const float* __restrict__ adj,
                   const float* __restrict__ w1, const float* __restrict__ b1,
                   const float* __restrict__ w2, const float* __restrict__ b2,
                   const float* __restrict__ n1g, const float* __restrict__ n1b,
                   const float* __restrict__ n2g, const float* __restrict__ n2b) {
  __shared__ __attribute__((aligned(32))) bf16t sA[64 * 64];  // Xb / src(bf16)
  __shared__ __attribute__((aligned(32))) bf16t sB[64 * 64];  // XbT / w1T / w2T
  __shared__ __attribute__((aligned(32))) bf16t sC[64 * 64];  // attn / h1
  __shared__ float sF[64 * 64];                                // S / src(f32)
  __shared__ float sG[64 * 64];                                // pre-LN2
  const int tid = threadIdx.x;
  const int w = tid >> 5, lane = tid & 31, hi = lane >> 4, cl = lane & 15;
  float* xc = x + (size_t)blockIdx.x * 64 * 64;
  bf16t* xbc = xb + (size_t)blockIdx.x * 64 * 64;
  const float* adc = adj + (size_t)blockIdx.x * 64 * 64;

  // phase 1: stage X (bf16) and X^T
  for (int e = tid; e < 4096; e += 256) {
    int rr = e >> 6, cc = e & 63;
    float v = xc[e];
    sA[e] = (bf16t)v;
    sB[cc * 64 + rr] = (bf16t)v;
  }
  __syncthreads();

  // phase 2: S = (X X^T) * scale * adj -> sF
  for (int t = w; t < 16; t += 8) {
    int mt = t >> 2, nt = t & 3;
    v8f acc = vzero8();
#pragma unroll
    for (int kt = 0; kt < 2; ++kt) {
      v16bf a = load_a_frag(sA, mt * 16, kt * 32, 64);
      v16bf b = load_bT_frag(sA, nt * 16, kt * 32, 64);
      acc = wmma_bf16(a, b, acc);
    }
    int col = nt * 16 + cl;
#pragma unroll
    for (int r = 0; r < 8; ++r) {
      int row = mt * 16 + r + 8 * hi;
      sF[row * 64 + col] = acc[r] * 0.125f * adc[row * 64 + col];
    }
  }
  __syncthreads();

  // phase 3: row softmax -> attn (bf16) in sC
  if (tid < 64) {
    const float* rowp = sF + tid * 64;
    float m = -1e30f;
    for (int j = 0; j < 64; ++j) m = fmaxf(m, rowp[j]);
    float s = 0.f;
    for (int j = 0; j < 64; ++j) s += expf(rowp[j] - m);
    float inv = 1.f / s;
    bf16t* op = sC + tid * 64;
    for (int j = 0; j < 64; ++j) op[j] = (bf16t)(expf(rowp[j] - m) * inv);
  }
  __syncthreads();

  // phase 4: O = attn @ X + x -> sF
  for (int t = w; t < 16; t += 8) {
    int mt = t >> 2, nt = t & 3;
    v8f acc = vzero8();
#pragma unroll
    for (int kt = 0; kt < 2; ++kt) {
      v16bf a = load_a_frag(sC, mt * 16, kt * 32, 64);
      v16bf b = load_bT_frag(sB, nt * 16, kt * 32, 64);
      acc = wmma_bf16(a, b, acc);
    }
    int col = nt * 16 + cl;
#pragma unroll
    for (int r = 0; r < 8; ++r) {
      int row = mt * 16 + r + 8 * hi;
      sF[row * 64 + col] = acc[r] + xc[row * 64 + col];
    }
  }
  __syncthreads();

  // phase 5: LN1 rows -> sF (f32) + sA (bf16); stage w1^T into sB
  if (tid < 64) {
    float* rowp = sF + tid * 64;
    float m = 0.f;
    for (int j = 0; j < 64; ++j) m += rowp[j];
    m *= (1.f / 64.f);
    float v = 0.f;
    for (int j = 0; j < 64; ++j) { float d = rowp[j] - m; v += d * d; }
    v *= (1.f / 64.f);
    float inv = rsqrtf(v + 1e-5f);
    for (int j = 0; j < 64; ++j) {
      float y = (rowp[j] - m) * inv * n1g[j] + n1b[j];
      rowp[j] = y;
      sA[tid * 64 + j] = (bf16t)y;
    }
  }
  for (int e = tid; e < 4096; e += 256) {
    int k = e >> 6, n = e & 63;
    sB[n * 64 + k] = (bf16t)w1[e];
  }
  __syncthreads();

  // phase 6: h1 = relu(src @ w1 + b1) -> sC (bf16)
  for (int t = w; t < 16; t += 8) {
    int mt = t >> 2, nt = t & 3;
    v8f acc = vzero8();
#pragma unroll
    for (int kt = 0; kt < 2; ++kt) {
      v16bf a = load_a_frag(sA, mt * 16, kt * 32, 64);
      v16bf b = load_bT_frag(sB, nt * 16, kt * 32, 64);
      acc = wmma_bf16(a, b, acc);
    }
    int col = nt * 16 + cl;
    float bias = b1[col];
#pragma unroll
    for (int r = 0; r < 8; ++r) {
      int row = mt * 16 + r + 8 * hi;
      sC[row * 64 + col] = (bf16t)fmaxf(acc[r] + bias, 0.f);
    }
  }
  __syncthreads();

  // phase 7: stage w2^T
  for (int e = tid; e < 4096; e += 256) {
    int k = e >> 6, n = e & 63;
    sB[n * 64 + k] = (bf16t)w2[e];
  }
  __syncthreads();

  // phase 8: h = h1 @ w2 + b2 + src -> sG
  for (int t = w; t < 16; t += 8) {
    int mt = t >> 2, nt = t & 3;
    v8f acc = vzero8();
#pragma unroll
    for (int kt = 0; kt < 2; ++kt) {
      v16bf a = load_a_frag(sC, mt * 16, kt * 32, 64);
      v16bf b = load_bT_frag(sB, nt * 16, kt * 32, 64);
      acc = wmma_bf16(a, b, acc);
    }
    int col = nt * 16 + cl;
    float bias = b2[col];
#pragma unroll
    for (int r = 0; r < 8; ++r) {
      int row = mt * 16 + r + 8 * hi;
      sG[row * 64 + col] = acc[r] + bias + sF[row * 64 + col];
    }
  }
  __syncthreads();

  // phase 9: LN2 rows -> x (f32) and xb (bf16)
  if (tid < 64) {
    float* rowp = sG + tid * 64;
    float m = 0.f;
    for (int j = 0; j < 64; ++j) m += rowp[j];
    m *= (1.f / 64.f);
    float v = 0.f;
    for (int j = 0; j < 64; ++j) { float d = rowp[j] - m; v += d * d; }
    v *= (1.f / 64.f);
    float inv = rsqrtf(v + 1e-5f);
    for (int j = 0; j < 64; ++j) {
      float y = (rowp[j] - m) * inv * n2g[j] + n2b[j];
      xc[tid * 64 + j] = y;
      xbc[tid * 64 + j] = (bf16t)y;
    }
  }
}

// ---------------------------------------------------------------------------
// Stage one tile of rows16 rows of the fc "total" matrix:
//   cols   0..127 : self + neighbor bf16 rows, async global->LDS b128
//   cols 128..191 : nbr_fea (f32->bf16) + zero pad, via VALU/ds
// 256 threads; rows16 = 16 or 32 (tasks = rows16*16, multiple of 256).
// ---------------------------------------------------------------------------
static __device__ inline void stage_fc_tile(bf16t* sA, const bf16t* __restrict__ xb,
                                            const float* __restrict__ nbr_fea,
                                            const int* __restrict__ nbr_idx, int rb,
                                            int rows16) {
  const int tid = threadIdx.x;
  const int ntask = rows16 * 16;
  for (int task = tid; task < ntask; task += 256) {
    int row = task >> 4;
    int sub = task & 15;          // 0..7 self, 8..15 neighbor
    int li = sub & 7;
    int gr = rb + row;
    int src_row = (sub < 8) ? (gr / MNB) : nbr_idx[gr];
    const bf16t* gp = xb + (size_t)src_row * 64 + li * 8;
    unsigned lds = (unsigned)(uintptr_t)(sA + row * KFCP + ((sub < 8) ? 0 : 64) + li * 8);
    async_b128(lds, gp);
  }
  for (int e = tid; e < rows16 * 64; e += 256) {
    int row = e >> 6, col = e & 63;
    int gr = rb + row;
    float v = (col < NBRF) ? nbr_fea[(size_t)gr * NBRF + col] : 0.f;
    sA[row * KFCP + 128 + col] = (bf16t)v;
  }
  wait_async0();
}

// ---------------------------------------------------------------------------
// Gated-fc pass 1: gated = total @ fcw + fcb; per-column sum/sumsq (BN1 stats)
// ---------------------------------------------------------------------------
__global__ void __launch_bounds__(256)
fc_pass1(const bf16t* __restrict__ xb, const float* __restrict__ nbr_fea,
         const int* __restrict__ nbr_idx, const bf16t* __restrict__ bsw,
         const float* __restrict__ fcb, float* __restrict__ sum1,
         float* __restrict__ sq1) {
  __shared__ __attribute__((aligned(32))) bf16t sA[16 * KFCP];
  const int tid = threadIdx.x;
  const int w = tid >> 5, lane = tid & 31, cl = lane & 15;
  v16bf Bf[6];
#pragma unroll
  for (int kt = 0; kt < 6; ++kt) Bf[kt] = load_bsw(bsw, kt, w, 8);
  const float bias = fcb[w * 16 + cl];
  float ps = 0.f, pq = 0.f;
  const int base0 = blockIdx.x * 256;
  for (int rt = 0; rt < 16; ++rt) {
    __syncthreads();
    const int rb = base0 + rt * 16;
    stage_fc_tile(sA, xb, nbr_fea, nbr_idx, rb, 16);
    if (rt + 1 < 16 && tid < 16)
      __builtin_prefetch(nbr_fea + (size_t)(rb + 16 + tid) * NBRF, 0, 3);
    __syncthreads();
    v8f acc = vzero8();
#pragma unroll
    for (int kt = 0; kt < 6; ++kt) {
      v16bf a = load_a_frag(sA, 0, kt * 32, KFCP);
      acc = wmma_bf16(a, Bf[kt], acc);
    }
#pragma unroll
    for (int r = 0; r < 8; ++r) {
      float g = acc[r] + bias;
      ps += g;
      pq += g * g;
    }
  }
  ps += __shfl_xor(ps, 16);
  pq += __shfl_xor(pq, 16);
  if (lane < 16) {
    atomicAdd(&sum1[w * 16 + lane], ps);
    atomicAdd(&sq1[w * 16 + lane], pq);
  }
}

// ---------------------------------------------------------------------------
// Gated-fc pass 2: recompute gated, BN1 affine, sigmoid(filt)*softplus(core),
// neighbor-sum into nbr_sum via global f32 atomics.
// ---------------------------------------------------------------------------
__global__ void __launch_bounds__(256)
fc_pass2(const bf16t* __restrict__ xb, const float* __restrict__ nbr_fea,
         const int* __restrict__ nbr_idx, const bf16t* __restrict__ bsw,
         const float* __restrict__ fcb, const float* __restrict__ sum1,
         const float* __restrict__ sq1, const float* __restrict__ g1,
         const float* __restrict__ bt1, float* __restrict__ nbr_sum) {
  __shared__ __attribute__((aligned(32))) bf16t sA[32 * KFCP];
  const int tid = threadIdx.x;
  const int w = tid >> 5, lane = tid & 31, hi = lane >> 4, cl = lane & 15;
  const int rs = w >> 2, cp = w & 3;
  v16bf Bf0[6], Bf1[6];
#pragma unroll
  for (int kt = 0; kt < 6; ++kt) {
    Bf0[kt] = load_bsw(bsw, kt, cp, 8);
    Bf1[kt] = load_bsw(bsw, kt, cp + 4, 8);
  }
  const float invc = 1.f / (float)NROWS;
  const int cF = cp * 16 + cl, cC = cF + 64;
  float mF = sum1[cF] * invc, vF = sq1[cF] * invc - mF * mF;
  float mC = sum1[cC] * invc, vC = sq1[cC] * invc - mC * mC;
  const float AF = g1[cF] * rsqrtf(vF + 1e-5f), BF = bt1[cF] - mF * AF;
  const float AC = g1[cC] * rsqrtf(vC + 1e-5f), BC = bt1[cC] - mC * AC;
  const float biasF = fcb[cF], biasC = fcb[cC];
  const int base0 = blockIdx.x * 256;
  for (int it = 0; it < 8; ++it) {
    __syncthreads();
    const int rb = base0 + it * 32;
    stage_fc_tile(sA, xb, nbr_fea, nbr_idx, rb, 32);
    __syncthreads();
    v8f a0 = vzero8(), a1 = vzero8();
#pragma unroll
    for (int kt = 0; kt < 6; ++kt) {
      v16bf a = load_a_frag(sA, rs * 16, kt * 32, KFCP);
      a0 = wmma_bf16(a, Bf0[kt], a0);
      a1 = wmma_bf16(a, Bf1[kt], a1);
    }
#pragma unroll
    for (int r = 0; r < 8; ++r) {
      float gf = (a0[r] + biasF) * AF + BF;
      float gc = (a1[r] + biasC) * AC + BC;
      float val = sigmoidf_(gf) * softplusf(gc);
      int gr = rb + rs * 16 + r + 8 * hi;
      int atom = gr / MNB;
      atomicAdd(&nbr_sum[(size_t)atom * 64 + cF], val);
    }
  }
}

// ---------------------------------------------------------------------------
// BN2 stats over nbr_sum [32768,64]
// ---------------------------------------------------------------------------
__global__ void __launch_bounds__(256)
bn_stats64(const float* __restrict__ v, float* __restrict__ sum,
           float* __restrict__ sq) {
  __shared__ float ls[256], lq[256];
  const int tid = threadIdx.x;
  const int c = tid & 63, rl = tid >> 6;
  float s = 0.f, q = 0.f;
  const int r1 = blockIdx.x * 512 + 512;
  for (int r = blockIdx.x * 512 + rl; r < r1; r += 4) {
    float t = v[(size_t)r * 64 + c];
    s += t;
    q += t * t;
  }
  ls[tid] = s;
  lq[tid] = q;
  __syncthreads();
  if (tid < 64) {
    s = ls[tid] + ls[tid + 64] + ls[tid + 128] + ls[tid + 192];
    q = lq[tid] + lq[tid + 64] + lq[tid + 128] + lq[tid + 192];
    atomicAdd(&sum[tid], s);
    atomicAdd(&sq[tid], q);
  }
}

__global__ void __launch_bounds__(256)
finalize_kernel(float* __restrict__ x, const float* __restrict__ ns,
                const float* __restrict__ sum2, const float* __restrict__ sq2,
                const float* __restrict__ g, const float* __restrict__ b) {
  int i = blockIdx.x * 256 + threadIdx.x;
  int c = i & 63;
  const float invn = 1.f / (float)N_ATOMS;
  float m = sum2[c] * invn;
  float var = sq2[c] * invn - m * m;
  float inv = rsqrtf(var + 1e-5f);
  float y = (ns[i] - m) * inv * g[c] + b[c];
  x[i] = softplusf(x[i] + y);
}

__global__ void __launch_bounds__(256) zero_kernel(float* __restrict__ p, int n) {
  int i = blockIdx.x * 256 + threadIdx.x;
  if (i < n) p[i] = 0.f;
}

__global__ void __launch_bounds__(256)
copy_kernel(const float* __restrict__ src, float* __restrict__ dst, int n) {
  int i = blockIdx.x * 256 + threadIdx.x;
  if (i < n) dst[i] = src[i];
}

// ---------------------------------------------------------------------------
struct ConvPtrs {
  const float *w1, *b1, *w2, *b2, *fcw, *fcb;
  const float *n1g, *n1b, *n2g, *n2b, *bn1g, *bn1b, *bn2g, *bn2b;
};

extern "C" void kernel_launch(void* const* d_in, const int* in_sizes, int n_in,
                              void* d_out, int out_size, void* d_ws, size_t ws_size,
                              hipStream_t stream) {
  const float* atom_fea = (const float*)d_in[0];
  const float* nbr_fea  = (const float*)d_in[1];
  const int*   nbr_idx  = (const int*)d_in[2];
  // d_in[3] = crystal_atom_idx (arange partition, not needed)
  const float* adj      = (const float*)d_in[4];

  // Detect params flattening convention at index 5:
  //   insertion order: emb_w (5888) first; sorted-key order: conv0.b1 (64) first.
  const float* emb_w;
  const float* emb_b;
  ConvPtrs cv[3];
  const int PB = 5;
  if (n_in > PB && in_sizes[PB] == 92 * 64) {
    emb_w = (const float*)d_in[PB + 0];
    emb_b = (const float*)d_in[PB + 1];
    for (int i = 0; i < 3; ++i) {
      const int o = PB + 2 + i * 14;
      cv[i].w1   = (const float*)d_in[o + 0];
      cv[i].b1   = (const float*)d_in[o + 1];
      cv[i].w2   = (const float*)d_in[o + 2];
      cv[i].b2   = (const float*)d_in[o + 3];
      cv[i].fcw  = (const float*)d_in[o + 4];
      cv[i].fcb  = (const float*)d_in[o + 5];
      cv[i].n1g  = (const float*)d_in[o + 6];
      cv[i].n1b  = (const float*)d_in[o + 7];
      cv[i].n2g  = (const float*)d_in[o + 8];
      cv[i].n2b  = (const float*)d_in[o + 9];
      cv[i].bn1g = (const float*)d_in[o + 10];
      cv[i].bn1b = (const float*)d_in[o + 11];
      cv[i].bn2g = (const float*)d_in[o + 12];
      cv[i].bn2b = (const float*)d_in[o + 13];
    }
  } else {
    // jax tree_flatten order: sorted keys; convs first, each conv sorted:
    // b1,b2,bn1b,bn1g,bn2b,bn2g,fcb,fcw,n1b,n1g,n2b,n2g,w1,w2; then emb_b, emb_w
    for (int i = 0; i < 3; ++i) {
      const int o = PB + i * 14;
      cv[i].b1   = (const float*)d_in[o + 0];
      cv[i].b2   = (const float*)d_in[o + 1];
      cv[i].bn1b = (const float*)d_in[o + 2];
      cv[i].bn1g = (const float*)d_in[o + 3];
      cv[i].bn2b = (const float*)d_in[o + 4];
      cv[i].bn2g = (const float*)d_in[o + 5];
      cv[i].fcb  = (const float*)d_in[o + 6];
      cv[i].fcw  = (const float*)d_in[o + 7];
      cv[i].n1b  = (const float*)d_in[o + 8];
      cv[i].n1g  = (const float*)d_in[o + 9];
      cv[i].n2b  = (const float*)d_in[o + 10];
      cv[i].n2g  = (const float*)d_in[o + 11];
      cv[i].w1   = (const float*)d_in[o + 12];
      cv[i].w2   = (const float*)d_in[o + 13];
    }
    emb_b = (const float*)d_in[PB + 42];
    emb_w = (const float*)d_in[PB + 43];
  }

  // workspace layout
  float* x       = (float*)d_ws;                    // N_ATOMS*64 f32
  float* nbr_sum = x + (size_t)N_ATOMS * 64;        // N_ATOMS*64 f32
  float* stats   = nbr_sum + (size_t)N_ATOMS * 64;  // 384 f32
  float* sum1 = stats, *sq1 = stats + 128, *sum2 = stats + 256, *sq2 = stats + 320;
  bf16t* bswE = (bf16t*)(stats + 384);              // 6144 bf16 (32B aligned)
  bf16t* bswF[3];
  bswF[0] = bswE + 6144;
  bswF[1] = bswF[0] + 6 * 8 * 512;
  bswF[2] = bswF[1] + 6 * 8 * 512;
  bf16t* xb = bswF[2] + 6 * 8 * 512;                // N_ATOMS*64 bf16 mirror

  // pre-swizzle weights to WMMA B-fragment layout (bf16)
  swizzleB_kernel<<<(3 * 4 * 512 + 255) / 256, 256, 0, stream>>>(emb_w, bswE, 92, 64, 3, 4);
  for (int i = 0; i < 3; ++i)
    swizzleB_kernel<<<(6 * 8 * 512 + 255) / 256, 256, 0, stream>>>(cv[i].fcw, bswF[i],
                                                                   KFC, NFC, 6, 8);

  // embedding
  embed_kernel<<<N_ATOMS / 32, 256, 0, stream>>>(atom_fea, bswE, emb_b, x);

  // conv layers
  for (int i = 0; i < 3; ++i) {
    transformer_kernel<<<N_CRY, 256, 0, stream>>>(x, xb, adj, cv[i].w1, cv[i].b1,
                                                  cv[i].w2, cv[i].b2, cv[i].n1g,
                                                  cv[i].n1b, cv[i].n2g, cv[i].n2b);
    const int nz = N_ATOMS * 64 + 384;  // nbr_sum + all stats (contiguous)
    zero_kernel<<<(nz + 255) / 256, 256, 0, stream>>>(nbr_sum, nz);
    fc_pass1<<<NROWS / 256, 256, 0, stream>>>(xb, nbr_fea, nbr_idx, bswF[i], cv[i].fcb,
                                              sum1, sq1);
    fc_pass2<<<NROWS / 256, 256, 0, stream>>>(xb, nbr_fea, nbr_idx, bswF[i], cv[i].fcb,
                                              sum1, sq1, cv[i].bn1g, cv[i].bn1b,
                                              nbr_sum);
    bn_stats64<<<64, 256, 0, stream>>>(nbr_sum, sum2, sq2);
    finalize_kernel<<<N_ATOMS * 64 / 256, 256, 0, stream>>>(x, nbr_sum, sum2, sq2,
                                                            cv[i].bn2g, cv[i].bn2b);
  }

  // out = x[crystal_atom_idx] == x reshaped [512,64,64]
  copy_kernel<<<(out_size + 255) / 256, 256, 0, stream>>>(x, (float*)d_out, out_size);
}
